// MSABlock_35983236006595
// MI455X (gfx1250) — matrix operand order repaired and checked by reading, not compiled, not run on hardware
//
#include <hip/hip_runtime.h>
#include <hip/hip_bf16.h>

// ---------------------------------------------------------------------------
// MSA block forward for MI455X (gfx1250, wave32, WMMA).
//  - Dense GEMMs: v_wmma_f32_16x16x32_bf16, 64x16 tile per wave (B-fragment
//    reused by 4 WMMAs), branch-free templated K-loop, float2 A loads,
//    global_prefetch_b8 for the next K tile.
//  - Row attention: flash-style, v_wmma_f32_16x16x4_f32 (full fp32 path),
//    64-wide j-tiles (8 score WMMAs + 16 PV WMMAs per iteration, one
//    lane-reduction per row per 64 cols), LDS probs tile + s_wait_dscnt.
// Shapes: S=32 Np=512 N=768 Dm=512 Dmsa=64 Hmsa=8 dh=8 Hres=16 R=16 Dff=256
// ---------------------------------------------------------------------------

typedef __attribute__((ext_vector_type(16))) __bf16 v16bf;
typedef __attribute__((ext_vector_type(8)))  float  v8f;
typedef __attribute__((ext_vector_type(2)))  float  v2f;

#define DEV static __device__ __forceinline__

DEV float sigmoidf_(float x) { return 1.0f / (1.0f + __expf(-x)); }

// ---------------------------------------------------------------------------
// Generic strided GEMM: C[m,n] = scale * sum_k A[m,k]*B[k,n]  (bf16 WMMA).
// One wave computes a 64x16 output tile (4 m-subtiles share one B fragment).
// K multiple of 32; M multiple of 64; N multiple of 16.
// Fragment layouts as loaded here (A/B in contiguous k order, not the ISA table order):
//   A 16x32 bf16 : lane m=l&15, half=l>>4; vgpr v holds K = 2*v + 16*half
//   B 32x16 bf16 : lane n=l&15; vgpr v holds K = 2*v + 16*half
//   C/D 16x16 f32: lane n=l&15; vgpr v holds row m = v + 8*half
// ---------------------------------------------------------------------------
template <bool AGATHER, bool BGATHER, bool ACONTIG>
__global__ __launch_bounds__(32) void gemm_bf16_wmma(
    const float* __restrict__ A, long a_rs, long a_cs, const int* __restrict__ a_rows,
    const float* __restrict__ B, long b_rs, long b_cs, const int* __restrict__ b_rows,
    float* __restrict__ C, long c_rs, long c_cs, int K, float scale)
{
  const int lane = threadIdx.x & 31;
  const int l16  = lane & 15;
  const int half = lane >> 4;
  const long m0 = (long)blockIdx.y * 64;
  const long n0 = (long)blockIdx.x * 16;

  const float* __restrict__ Ab[4];
#pragma unroll
  for (int t = 0; t < 4; ++t) {
    const long am = m0 + 16 * t + l16;
    const long arow = AGATHER ? (long)a_rows[am] : am;
    Ab[t] = A + arow * a_rs;
  }
  const long bn = (n0 + l16) * b_cs;

  v8f acc[4] = {{}, {}, {}, {}};
  for (int k0 = 0; k0 < K; k0 += 32) {
    v16bf bf;
#pragma unroll
    for (int v = 0; v < 8; ++v) {
      const int kb = k0 + (v << 1) + (half << 4);
      const long r0 = (BGATHER ? (long)b_rows[kb]     : (long)kb)       * b_rs;
      const long r1 = (BGATHER ? (long)b_rows[kb + 1] : (long)(kb + 1)) * b_rs;
      bf[2 * v    ] = (__bf16)B[r0 + bn];
      bf[2 * v + 1] = (__bf16)B[r1 + bn];
    }
    if (!BGATHER) __builtin_prefetch(B + (long)(k0 + 32 + (half << 4)) * b_rs + bn, 0, 3);
#pragma unroll
    for (int t = 0; t < 4; ++t) {
      v16bf af;
#pragma unroll
      for (int v = 0; v < 8; ++v) {
        const int ka = k0 + (v << 1) + (half << 4);
        if (ACONTIG) {
          v2f p = *(const v2f*)(Ab[t] + ka);       // pairs contiguous, 8B aligned
          af[2 * v    ] = (__bf16)p[0];
          af[2 * v + 1] = (__bf16)p[1];
        } else {
          af[2 * v    ] = (__bf16)Ab[t][(long)ka * a_cs];
          af[2 * v + 1] = (__bf16)Ab[t][(long)(ka + 1) * a_cs];
        }
      }
      if (ACONTIG) __builtin_prefetch(Ab[t] + (k0 + 32 + (half << 3)), 0, 3);
      acc[t] = __builtin_amdgcn_wmma_f32_16x16x32_bf16(false, af, false, bf,
                                                       (short)0, acc[t], false, false);
    }
  }
#pragma unroll
  for (int t = 0; t < 4; ++t)
#pragma unroll
    for (int v = 0; v < 8; ++v) {
      long m = m0 + 16 * t + v + (half << 3);
      C[m * c_rs + (n0 + l16) * c_cs] = acc[t][v] * scale;
    }
}

// ---------------------------------------------------------------------------
// Flash row-attention, one wave per (s, h, 16-row tile). d_head = 8.
// 64-wide j-tiles; QK^T and PV via v_wmma_f32_16x16x4_f32 (fp32 WMMA).
// f32 A 16x4: lane m=l&15; vgpr v holds K = v + 2*half. B 4x16 mirrored.
// ---------------------------------------------------------------------------
__global__ __launch_bounds__(32) void attn_flash_f32wmma(
    const float* __restrict__ Q, const float* __restrict__ Km,
    const float* __restrict__ Vm, const float* __restrict__ pb,
    float* __restrict__ O, int Np)
{
  __shared__ float sP[16 * 64];        // 16x64 probs tile
  const int lane = threadIdx.x & 31;
  const int l16 = lane & 15, half = lane >> 4;
  const int i0 = blockIdx.x * 16;
  const int h  = blockIdx.y;
  const int s  = blockIdx.z;
  const long rowbase = ((long)s * Np) * 64 + h * 8;   // [s, row, h*8 + d]

  const float* qrow = Q + rowbase + (long)(i0 + l16) * 64;
  v2f aq0, aq1;
  aq0[0] = qrow[0 + 2 * half]; aq0[1] = qrow[1 + 2 * half];
  aq1[0] = qrow[4 + 2 * half]; aq1[1] = qrow[5 + 2 * half];

  v8f o = {};
  float mrow[8], lrow[8];
#pragma unroll
  for (int v = 0; v < 8; ++v) { mrow[v] = -1e30f; lrow[v] = 0.0f; }
  const float qscale = 0.3535533905932738f;   // 8^-0.5

  for (int j0 = 0; j0 < Np; j0 += 64) {
    // four 16x16 score tiles = Q(16x8) @ K^T(8x64), + pos_bias
    v8f sc[4];
#pragma unroll
    for (int t = 0; t < 4; ++t) {
      const float* krow = Km + rowbase + (long)(j0 + 16 * t + l16) * 64;
      v2f bk0, bk1;
      bk0[0] = krow[0 + 2 * half]; bk0[1] = krow[1 + 2 * half];
      bk1[0] = krow[4 + 2 * half]; bk1[1] = krow[5 + 2 * half];
      v8f sr = {};
      sr = __builtin_amdgcn_wmma_f32_16x16x4_f32(false, aq0, false, bk0, (short)0, sr, false, false);
      sr = __builtin_amdgcn_wmma_f32_16x16x4_f32(false, aq1, false, bk1, (short)0, sr, false, false);
#pragma unroll
      for (int v = 0; v < 8; ++v)
        sc[t][v] = sr[v] * qscale +
                   pb[((long)h * Np + (i0 + v + 8 * half)) * (long)Np + j0 + 16 * t + l16];
    }

    // online softmax: one lane-reduction per row per 64 columns
#pragma unroll
    for (int v = 0; v < 8; ++v) {
      const int m = v + 8 * half;
      float t0 = fmaxf(fmaxf(sc[0][v], sc[1][v]), fmaxf(sc[2][v], sc[3][v]));
      t0 = fmaxf(t0, __shfl_xor(t0, 1));
      t0 = fmaxf(t0, __shfl_xor(t0, 2));
      t0 = fmaxf(t0, __shfl_xor(t0, 4));
      t0 = fmaxf(t0, __shfl_xor(t0, 8));
      float mnew = fmaxf(mrow[v], t0);
      float corr = __expf(mrow[v] - mnew);
      float ps = 0.f;
#pragma unroll
      for (int t = 0; t < 4; ++t) {
        float p = __expf(sc[t][v] - mnew);
        sP[m * 64 + 16 * t + l16] = p;
        ps += p;
      }
      ps += __shfl_xor(ps, 1);
      ps += __shfl_xor(ps, 2);
      ps += __shfl_xor(ps, 4);
      ps += __shfl_xor(ps, 8);
      lrow[v] = lrow[v] * corr + ps;
      mrow[v] = mnew;
      o[v] *= corr;
    }
    asm volatile("s_wait_dscnt 0x0" ::: "memory");  // LDS writes visible (same wave)

    // o += P(16x64) @ V(64x8, padded to 16 cols)
#pragma unroll
    for (int kb = 0; kb < 64; kb += 4) {
      v2f ap, bv;
      ap[0] = sP[l16 * 64 + kb + 0 + 2 * half];
      ap[1] = sP[l16 * 64 + kb + 1 + 2 * half];
      int k0r = j0 + kb + 2 * half;
      const float* v0 = Vm + rowbase + (long)(k0r) * 64;
      const float* v1 = Vm + rowbase + (long)(k0r + 1) * 64;
      bv[0] = (l16 < 8) ? v0[l16] : 0.0f;
      bv[1] = (l16 < 8) ? v1[l16] : 0.0f;
      o = __builtin_amdgcn_wmma_f32_16x16x4_f32(false, ap, false, bv, (short)0, o, false, false);
    }
    asm volatile("s_wait_dscnt 0x0" ::: "memory");
  }
#pragma unroll
  for (int v = 0; v < 8; ++v) {
    if (l16 < 8)
      O[rowbase + (long)(i0 + v + 8 * half) * 64 + l16] = o[v] / lrow[v];
  }
}

// ------------------------------ VALU kernels --------------------------------

__global__ __launch_bounds__(32) void layernorm_rows(
    const float* __restrict__ x, const float* __restrict__ g,
    const float* __restrict__ b, float* __restrict__ y, int D)
{
  const int lane = threadIdx.x & 31;
  const long row = blockIdx.x;
  const float* xr = x + row * D;
  float s = 0.f, s2 = 0.f;
  for (int d = lane; d < D; d += 32) { float v = xr[d]; s += v; s2 += v * v; }
  for (int off = 16; off; off >>= 1) { s += __shfl_xor(s, off); s2 += __shfl_xor(s2, off); }
  float mu  = s / D;
  float var = s2 / D - mu * mu;
  float inv = rsqrtf(var + 1e-5f);
  float* yr = y + row * D;
  for (int d = lane; d < D; d += 32) yr[d] = (xr[d] - mu) * inv * g[d] + b[d];
}

__global__ __launch_bounds__(256) void add_inject(
    const float* __restrict__ m, const float* __restrict__ t,
    const float* __restrict__ bias, float* __restrict__ out, int Np, long n)
{
  long i = (long)blockIdx.x * blockDim.x + threadIdx.x;
  if (i >= n) return;
  int  d   = (int)(i & 63);
  long row = (i >> 6) % Np;
  out[i] = m[i] + t[row * 64 + d] + bias[d];
}

__global__ __launch_bounds__(256) void copy_f32(const float* a, float* o, long n)
{ long i = (long)blockIdx.x * blockDim.x + threadIdx.x; if (i < n) o[i] = a[i]; }

__global__ __launch_bounds__(256) void zero_f32(float* o, long n)
{ long i = (long)blockIdx.x * blockDim.x + threadIdx.x; if (i < n) o[i] = 0.0f; }

__global__ __launch_bounds__(256) void gate_update(
    float* __restrict__ m, const float* __restrict__ G,
    const float* __restrict__ ap, long n)
{
  long i = (long)blockIdx.x * blockDim.x + threadIdx.x;
  if (i < n) m[i] += sigmoidf_(G[i]) * ap[i];
}

__global__ __launch_bounds__(256) void silu_mul(
    float* __restrict__ f1, const float* __restrict__ f2, long n)
{
  long i = (long)blockIdx.x * blockDim.x + threadIdx.x;
  if (i < n) { float x = f1[i]; f1[i] = (x * sigmoidf_(x)) * f2[i]; }
}

__global__ __launch_bounds__(256) void add_inplace(float* o, const float* a, long n)
{ long i = (long)blockIdx.x * blockDim.x + threadIdx.x; if (i < n) o[i] += a[i]; }

__global__ __launch_bounds__(256) void add_bias_rows(float* o, const float* b, int D, long n)
{ long i = (long)blockIdx.x * blockDim.x + threadIdx.x; if (i < n) o[i] += b[i % D]; }

// softmax over S (lanes = sequences) of mn @ w_colw
__global__ __launch_bounds__(32) void col_alpha(
    const float* __restrict__ mn, const float* __restrict__ wcol,
    float* __restrict__ alpha, int Np)
{
  int i = blockIdx.x, s = threadIdx.x;
  const float* xr = mn + ((long)s * Np + i) * 64;
  float z = 0.f;
  for (int d = 0; d < 64; ++d) z += xr[d] * wcol[d];
  float zm = z;
  for (int off = 16; off; off >>= 1) zm = fmaxf(zm, __shfl_xor(zm, off));
  float e = __expf(z - zm);
  float se = e;
  for (int off = 16; off; off >>= 1) se += __shfl_xor(se, off);
  alpha[(long)s * Np + i] = e / se;
}

__global__ __launch_bounds__(32) void col_agg_k(
    const float* __restrict__ mn, const float* __restrict__ alpha,
    float* __restrict__ agg, int Np, int S)
{
  int i = blockIdx.x, lane = threadIdx.x;
  for (int d = lane; d < 64; d += 32) {
    float acc = 0.f;
    for (int s = 0; s < S; ++s)
      acc += alpha[(long)s * Np + i] * mn[((long)s * Np + i) * 64 + d];
    agg[(long)i * 64 + d] = acc;
  }
}

__global__ __launch_bounds__(256) void scatter_add_rows(
    float* __restrict__ out, const float* __restrict__ upd,
    const int* __restrict__ idx, int D)
{
  int i = blockIdx.x;
  int r = idx[i];
  for (int d = threadIdx.x; d < D; d += blockDim.x)
    atomicAdd(&out[(long)r * D + d], upd[(long)i * D + d]);
}

// w[i,j] = sigmoid( sum_r c[i,j,r]*w_cw[r] + b_cw )
__global__ __launch_bounds__(256) void coevol_gate(
    const float* __restrict__ c, const float* __restrict__ wcw,
    const float* __restrict__ bcw, float* __restrict__ w, long n)
{
  long t = (long)blockIdx.x * blockDim.x + threadIdx.x;
  if (t >= n) return;
  const float* cr = c + t * 16;
  float acc = bcw[0];
#pragma unroll
  for (int r = 0; r < 16; ++r) acc += cr[r] * wcw[r];
  w[t] = sigmoidf_(acc);
}

// cbias[idx[i],h] = sum_r (c.sum(1)[i,r]/Np) * w_c2m[r,h] + b_c2m[h]
__global__ __launch_bounds__(32) void coevol_marg(
    const float* __restrict__ c, const float* __restrict__ w_c2m,
    const float* __restrict__ b_c2m, const int* __restrict__ idx,
    float* __restrict__ cbias, int Np)
{
  __shared__ float sb[16];
  int i = blockIdx.x, lane = threadIdx.x;
  if (lane < 16) {
    float s = 0.f;
    const float* cp = c + (long)i * Np * 16 + lane;
    for (int j = 0; j < Np; ++j) s += cp[(long)j * 16];
    sb[lane] = s / (float)Np;
  }
  __syncthreads();
  if (lane < 16) {
    float acc = b_c2m[lane];
#pragma unroll
    for (int r = 0; r < 16; ++r) acc += sb[r] * w_c2m[r * 16 + lane];
    cbias[(long)idx[i] * 16 + lane] = acc;
  }
}

// out[hh,:] = softmax_n( logit[n,hh] + bias[hh] + aco[hh]*cbias[n,hh] )
__global__ __launch_bounds__(32) void head_softmax(
    const float* __restrict__ logit, const float* __restrict__ bias,
    const float* __restrict__ aco, const float* __restrict__ cbias,
    float* __restrict__ out, int N, int H)
{
  int hh = blockIdx.x, lane = threadIdx.x;
  float bh = bias[hh], ah = aco[hh];
  float mx = -1e30f;
  for (int n = lane; n < N; n += 32)
    mx = fmaxf(mx, logit[(long)n * H + hh] + bh + ah * cbias[(long)n * H + hh]);
  for (int off = 16; off; off >>= 1) mx = fmaxf(mx, __shfl_xor(mx, off));
  float se = 0.f;
  for (int n = lane; n < N; n += 32)
    se += __expf(logit[(long)n * H + hh] + bh + ah * cbias[(long)n * H + hh] - mx);
  for (int off = 16; off; off >>= 1) se += __shfl_xor(se, off);
  float inv = 1.0f / se;
  for (int n = lane; n < N; n += 32) {
    float v = logit[(long)n * H + hh] + bh + ah * cbias[(long)n * H + hh];
    out[(long)hh * N + n] = __expf(v - mx) * inv;
  }
}

// ------------------------------- host side ---------------------------------

static inline void gemm(hipStream_t st,
                        const float* A, long ars, long acs, const int* arows,
                        const float* B, long brs, long bcs, const int* brows,
                        float* C, long crs, long ccs, int M, int N, int K, float scale)
{
  dim3 g((unsigned)(N / 16), (unsigned)(M / 64)), b(32);
  if (arows)
    gemm_bf16_wmma<true, false, true><<<g, b, 0, st>>>(A, ars, acs, arows, B, brs, bcs, brows, C, crs, ccs, K, scale);
  else if (brows)
    gemm_bf16_wmma<false, true, true><<<g, b, 0, st>>>(A, ars, acs, arows, B, brs, bcs, brows, C, crs, ccs, K, scale);
  else if (acs == 1)
    gemm_bf16_wmma<false, false, true><<<g, b, 0, st>>>(A, ars, acs, arows, B, brs, bcs, brows, C, crs, ccs, K, scale);
  else
    gemm_bf16_wmma<false, false, false><<<g, b, 0, st>>>(A, ars, acs, arows, B, brs, bcs, brows, C, crs, ccs, K, scale);
}

extern "C" void kernel_launch(void* const* d_in, const int* in_sizes, int n_in,
                              void* d_out, int out_size, void* d_ws, size_t ws_size,
                              hipStream_t stream)
{
  (void)in_sizes; (void)n_in; (void)out_size; (void)ws_size;
  constexpr int  S = 32, NP = 512, NR = 768, DM = 512, DMSA = 64, H = 8,
                 HR = 16, DFF = 256;
  constexpr long MROWS = (long)S * NP;                // 16384
  constexpr long MELEM = MROWS * DMSA;                // 1048576

  const float* m_in  = (const float*)d_in[0];
  const float* h_in  = (const float*)d_in[1];
  const float* pos_b = (const float*)d_in[4];
  const float* a_co  = (const float*)d_in[5];
  const int*   idx   = (const int*)  d_in[6];
  void* const* P = d_in + 7;   // params, dict insertion order
  const float* w_s2m = (const float*)P[0],  *b_s2m = (const float*)P[1];
  const float* lrg   = (const float*)P[2],  *lrb   = (const float*)P[3];
  const float* w_q   = (const float*)P[4],  *w_k   = (const float*)P[5];
  const float* w_v   = (const float*)P[6],  *w_g   = (const float*)P[7];
  const float* w_o   = (const float*)P[8];
  const float* lcg   = (const float*)P[9],  *lcb   = (const float*)P[10];
  const float* w_colw= (const float*)P[11], *w_c2s = (const float*)P[12];
  const float* log_  = (const float*)P[13], *lob   = (const float*)P[14];
  const float* w_u   = (const float*)P[15], *w_v2  = (const float*)P[16];
  const float* cvg   = (const float*)P[17], *cvb   = (const float*)P[18];
  const float* cv_w  = (const float*)P[19], *cv_b  = (const float*)P[20];
  const float* w_cw  = (const float*)P[21], *b_cw  = (const float*)P[22];
  const float* w_co  = (const float*)P[23], *b_co  = (const float*)P[24];
  const float* w_mu  = (const float*)P[25], *b_mu  = (const float*)P[26];
  const float* w_nu  = (const float*)P[27], *b_nu  = (const float*)P[28];
  const float* w_c2m = (const float*)P[29], *b_c2m = (const float*)P[30];
  const float* lfg   = (const float*)P[31], *lfb   = (const float*)P[32];
  const float* w_ff1 = (const float*)P[33], *w_ff2 = (const float*)P[34];
  const float* w_ff3 = (const float*)P[35];

  float* ws = (float*)d_ws;
  float* mn   = ws;
  float* Qb   = ws + 1 * MELEM;
  float* Kb   = ws + 2 * MELEM;
  float* Vb   = ws + 3 * MELEM;
  float* Gb   = ws + 4 * MELEM;
  float* attO = ws + 5 * MELEM;
  float* attP = ws + 6 * MELEM;
  size_t o = 7 * (size_t)MELEM;
  float* tinj  = ws + o; o += (size_t)NP * DMSA;     // 512x64
  float* alpha = ws + o; o += (size_t)S * NP;        // 32x512
  float* cagg  = ws + o; o += (size_t)NP * DMSA;     // 512x64
  float* upd   = ws + o; o += (size_t)NP * DM;       // 512x512
  float* Ub    = ws + o; o += (size_t)S * NP * HR;   // 32x512x16
  float* Vc    = ws + o; o += (size_t)S * NP * HR;
  float* hc    = ws + o; o += (size_t)NR * DM;       // 768x512
  float* cbuf  = ws + o; o += (size_t)NP * NP * HR;  // 512x512x16
  float* wmat  = ws + o; o += (size_t)NP * NP;       // 512x512
  float* g2    = ws + o; o += (size_t)NP * DM;
  float* g3    = ws + o; o += (size_t)NP * DM;
  float* mul_  = ws + o; o += (size_t)NR * HR;
  float* nul_  = ws + o; o += (size_t)NR * HR;
  float* cbias = ws + o; o += (size_t)NR * HR;
  float* f1  = Qb;    // Q..G region = 4*MELEM = 16384*256 floats exactly
  float* f2  = cbuf;  // c region    = 512*512*16 = 16384*256 floats exactly
  float* ffo = attO;

  float* out_m  = (float*)d_out;
  float* out_h  = out_m + MELEM;
  float* out_mu = out_h + (long)NR * DM;
  float* out_nu = out_mu + (long)HR * NR;

  auto ew = [](long n) { return dim3((unsigned)((n + 255) / 256)); };

  // 1) inject single repr: m1 = m + (h_res[idx] @ w_s2m + b_s2m)
  gemm(stream, h_in, DM, 1, idx, w_s2m, DMSA, 1, nullptr, tinj, DMSA, 1, NP, DMSA, DM, 1.0f);
  add_inject<<<ew(MELEM), 256, 0, stream>>>(m_in, tinj, b_s2m, out_m, NP, MELEM);

  // 2) row attention
  layernorm_rows<<<dim3((unsigned)MROWS), 32, 0, stream>>>(out_m, lrg, lrb, mn, DMSA);
  gemm(stream, mn, DMSA, 1, nullptr, w_q, DMSA, 1, nullptr, Qb, DMSA, 1, (int)MROWS, DMSA, DMSA, 1.0f);
  gemm(stream, mn, DMSA, 1, nullptr, w_k, DMSA, 1, nullptr, Kb, DMSA, 1, (int)MROWS, DMSA, DMSA, 1.0f);
  gemm(stream, mn, DMSA, 1, nullptr, w_v, DMSA, 1, nullptr, Vb, DMSA, 1, (int)MROWS, DMSA, DMSA, 1.0f);
  gemm(stream, mn, DMSA, 1, nullptr, w_g, DMSA, 1, nullptr, Gb, DMSA, 1, (int)MROWS, DMSA, DMSA, 1.0f);
  attn_flash_f32wmma<<<dim3(NP / 16, H, S), 32, 0, stream>>>(Qb, Kb, Vb, pos_b, attO, NP);
  gemm(stream, attO, DMSA, 1, nullptr, w_o, DMSA, 1, nullptr, attP, DMSA, 1, (int)MROWS, DMSA, DMSA, 1.0f);
  gate_update<<<ew(MELEM), 256, 0, stream>>>(out_m, Gb, attP, MELEM);   // -> m2

  // 3) column pooling -> h_res update
  copy_f32<<<ew((long)NR * DM), 256, 0, stream>>>(h_in, out_h, (long)NR * DM);
  layernorm_rows<<<dim3((unsigned)MROWS), 32, 0, stream>>>(out_m, lcg, lcb, mn, DMSA);
  col_alpha<<<dim3(NP), 32, 0, stream>>>(mn, w_colw, alpha, NP);
  col_agg_k<<<dim3(NP), 32, 0, stream>>>(mn, alpha, cagg, NP, S);
  gemm(stream, cagg, DMSA, 1, nullptr, w_c2s, DM, 1, nullptr, upd, DM, 1, NP, DM, DMSA, 1.0f);
  scatter_add_rows<<<dim3(NP), 256, 0, stream>>>(out_h, upd, idx, DM);

  // 4) low-rank coevolution
  layernorm_rows<<<dim3((unsigned)MROWS), 32, 0, stream>>>(out_m, log_, lob, mn, DMSA);
  gemm(stream, mn, DMSA, 1, nullptr, w_u,  HR, 1, nullptr, Ub, HR, 1, (int)MROWS, HR, DMSA, 1.0f);
  gemm(stream, mn, DMSA, 1, nullptr, w_v2, HR, 1, nullptr, Vc, HR, 1, (int)MROWS, HR, DMSA, 1.0f);
  layernorm_rows<<<dim3(NR), 32, 0, stream>>>(out_h, cvg, cvb, mn, DM);
  gemm(stream, mn, DM, 1, nullptr, cv_w, DM, 1, nullptr, hc, DM, 1, NR, DM, DM, 1.0f);
  add_bias_rows<<<ew((long)NR * DM), 256, 0, stream>>>(hc, cv_b, DM, (long)NR * DM);
  // c[i,j,r] = sum_s U[s,i,r] Vc[s,j,r] / S : 16 strided GEMMs (M=N=512, K=32)
  for (int r = 0; r < HR; ++r)
    gemm(stream, Ub + r, 16, (long)NP * HR, nullptr,
                 Vc + r, (long)NP * HR, 16, nullptr,
                 cbuf + r, (long)NP * HR, 16, NP, NP, S, 1.0f / (float)S);
  coevol_gate<<<ew((long)NP * NP), 256, 0, stream>>>(cbuf, w_cw, b_cw, wmat, (long)NP * NP);
  gemm(stream, wmat, NP, 1, nullptr, hc, DM, 1, idx, g2, DM, 1, NP, DM, NP, 1.0f);
  gemm(stream, g2, DM, 1, nullptr, w_co, DM, 1, nullptr, g3, DM, 1, NP, DM, DM, 1.0f);
  add_bias_rows<<<ew((long)NR * DM), 256, 0, stream>>>(out_h, b_co, DM, (long)NR * DM);
  scatter_add_rows<<<dim3(NP), 256, 0, stream>>>(out_h, g3, idx, DM);

  // 5) marginals
  zero_f32<<<ew((long)NR * HR), 256, 0, stream>>>(cbias, (long)NR * HR);
  coevol_marg<<<dim3(NP), 32, 0, stream>>>(cbuf, w_c2m, b_c2m, idx, cbias, NP);
  gemm(stream, out_h, DM, 1, nullptr, w_mu, HR, 1, nullptr, mul_, HR, 1, NR, HR, DM, 1.0f);
  gemm(stream, out_h, DM, 1, nullptr, w_nu, HR, 1, nullptr, nul_, HR, 1, NR, HR, DM, 1.0f);
  head_softmax<<<dim3(HR), 32, 0, stream>>>(mul_, b_mu, a_co, cbias, out_mu, NR, HR);
  head_softmax<<<dim3(HR), 32, 0, stream>>>(nul_, b_nu, a_co, cbias, out_nu, NR, HR);

  // 6) SwiGLU FFN (note: c-buffer and Q..G region reused as f2/f1)
  layernorm_rows<<<dim3((unsigned)MROWS), 32, 0, stream>>>(out_m, lfg, lfb, mn, DMSA);
  gemm(stream, mn, DMSA, 1, nullptr, w_ff1, DFF, 1, nullptr, f1, DFF, 1, (int)MROWS, DFF, DMSA, 1.0f);
  gemm(stream, mn, DMSA, 1, nullptr, w_ff2, DFF, 1, nullptr, f2, DFF, 1, (int)MROWS, DFF, DMSA, 1.0f);
  silu_mul<<<ew(MROWS * DFF), 256, 0, stream>>>(f1, f2, MROWS * DFF);
  gemm(stream, f1, DFF, 1, nullptr, w_ff3, DMSA, 1, nullptr, ffo, DMSA, 1, (int)MROWS, DMSA, DFF, 1.0f);
  add_inplace<<<ew(MELEM), 256, 0, stream>>>(out_m, ffo, MELEM);
}
